// HopfieldAssociativeMemory_84808424227418
// MI455X (gfx1250) — matrix operand order repaired and checked by reading, not compiled, not run on hardware
//
#include <hip/hip_runtime.h>
#include <hip/hip_bf16.h>
#include <math.h>

// ---------------------------------------------------------------------------
// HopfieldAssociativeMemory for MI455X (gfx1250, wave32, WMMA).
//
// Roofline: sims GEMM = 68.7 GFLOP over a 512MB pattern stream (128 FLOP/B).
// fp32 can't keep up (WMMA f32 is K=4); bf16 WMMA (K=32) matches the ~22us
// HBM stream, so: fp32->bf16 convert-on-the-fly into LDS, bf16 query in L2,
// v_wmma_f32_16x16x32_bf16 tiles. Extra sims round-trip (~512MB) adds ~22us.
// Top-k streaming pass uses GLOBAL_LOAD_ASYNC_TO_LDS_B128 (ASYNCcnt path,
// double-buffered with s_wait_asynccnt) per CDNA5 ISA.
//
// ws layout: [0, 256KB)        qn_bf16  (normalized query, bf16)
//            [1MB, 1MB+256MB)  sims     (fp32, 256 x 262144)
//            [1MB+256MB, +32K) top_idx  (int, 256 x 32)
// ---------------------------------------------------------------------------

// --- probes: report (non-fatally) which CDNA5 async/TDM builtins exist ----
#if !__has_builtin(__builtin_amdgcn_global_load_async_to_lds_b128)
#warning "PROBE: missing __builtin_amdgcn_global_load_async_to_lds_b128"
#endif
#if !__has_builtin(__builtin_amdgcn_s_wait_asynccnt)
#warning "PROBE: missing __builtin_amdgcn_s_wait_asynccnt"
#endif
#if !__has_builtin(__builtin_amdgcn_tensor_load_to_lds)
#warning "PROBE: missing __builtin_amdgcn_tensor_load_to_lds"
#endif

#define N_PAT 262144
#define BQ    256
#define DIM   512
#define NT    128      // patterns per workgroup tile
#define PSTRIDE 520    // padded LDS row stride (bf16 elems) to spread banks
#define TOPK  32

typedef __attribute__((ext_vector_type(16))) __bf16 v16bf;
typedef __attribute__((ext_vector_type(8)))  __bf16 v8bf;
typedef __attribute__((ext_vector_type(4)))  __bf16 v4bf;
typedef __attribute__((ext_vector_type(8)))  float  v8f;

// Per-lane async copy: 16B global -> 16B LDS, tracked by ASYNCcnt.
// ldsoff is the workgroup-relative LDS byte offset (low 32 bits of a generic
// __shared__ pointer), gaddr a per-lane 64-bit global address.
__device__ __forceinline__ void async_g2l_b128(unsigned ldsoff, unsigned long long gaddr) {
  asm volatile("global_load_async_to_lds_b128 %0, %1, off"
               :: "v"(ldsoff), "v"(gaddr) : "memory");
}
__device__ __forceinline__ void wait_asynccnt_le1() {
  asm volatile("s_wait_asynccnt 0x1" ::: "memory");
}
__device__ __forceinline__ void wait_asynccnt_0() {
  asm volatile("s_wait_asynccnt 0x0" ::: "memory");
}

// ---------------------------------------------------------------------------
// Kernel 1: qn = q / max(||q||, eps), stored as bf16 (stays hot in 192MB L2)
// ---------------------------------------------------------------------------
__global__ __launch_bounds__(256) void qnorm_kernel(const float* __restrict__ q,
                                                    __bf16* __restrict__ qnb) {
  __shared__ float red[8];
  const int t = threadIdx.x;
  const int row = blockIdx.x;
  const float a = q[(size_t)row * DIM + t];
  const float b = q[(size_t)row * DIM + 256 + t];
  float ss = a * a + b * b;
#pragma unroll
  for (int m = 16; m > 0; m >>= 1) ss += __shfl_xor(ss, m, 32);
  if ((t & 31) == 0) red[t >> 5] = ss;
  __syncthreads();
  float tot = 0.f;
#pragma unroll
  for (int i = 0; i < 8; ++i) tot += red[i];
  const float rn = 1.0f / fmaxf(sqrtf(tot), 1e-8f);
  qnb[(size_t)row * DIM + t] = (__bf16)(a * rn);
  qnb[(size_t)row * DIM + 256 + t] = (__bf16)(b * rn);
}

// ---------------------------------------------------------------------------
// Kernel 2: sims[256 x 262144] via bf16 WMMA.
// Grid: N_PAT/NT = 2048 workgroups x 256 threads (8 waves).
// Each WG: stage 128 patterns (fp32 -> bf16) into LDS + fp32 sumsq norms,
// then wave w computes M-blocks {2w, 2w+1} x 8 N-blocks, K-loop of 16 steps
// of v_wmma_f32_16x16x32_bf16 (B fragments from LDS, A fragments from L2).
// ---------------------------------------------------------------------------
__global__ __launch_bounds__(256) void sims_wmma_kernel(
    const float* __restrict__ mem, const __bf16* __restrict__ qnb,
    float* __restrict__ sims) {
  __shared__ __align__(16) __bf16 pB[NT][PSTRIDE];   // ~130 KB (of 320KB WGP LDS)
  __shared__ float part[256];
  __shared__ float rnorm[NT];

  const int t = threadIdx.x;
  const size_t pat0 = (size_t)blockIdx.x * NT;

  // ---- stage: fp32 patterns -> bf16 LDS, accumulate fp32 sum-of-squares ----
  {
    const int p = t >> 1, h = t & 1;  // 2 threads per pattern, 256 floats each
    const float* src = mem + (pat0 + p) * DIM + h * 256;
    float ss = 0.f;
#pragma unroll 4
    for (int j = 0; j < 64; ++j) {
      const float4 v = ((const float4*)src)[j];
      __builtin_prefetch(src + j * 4 + 64, 0, 3);  // global_prefetch_b8
      ss = fmaf(v.x, v.x, fmaf(v.y, v.y, fmaf(v.z, v.z, fmaf(v.w, v.w, ss))));
      v4bf o;
      o[0] = (__bf16)v.x; o[1] = (__bf16)v.y;
      o[2] = (__bf16)v.z; o[3] = (__bf16)v.w;
      *(v4bf*)&pB[p][h * 256 + j * 4] = o;
    }
    part[t] = ss;
  }
  __syncthreads();
  if (t < NT) rnorm[t] = 1.0f / fmaxf(sqrtf(part[2 * t] + part[2 * t + 1]), 1e-8f);
  __syncthreads();

  const int lane = t & 31;
  const int wave = t >> 5;
  const int nidx = lane & 15;   // N (or M) index within 16
  const int half = lane >> 4;   // lane-half selects K sub-range

  v8f acc[2][8];
#pragma unroll
  for (int i = 0; i < 2; ++i)
#pragma unroll
    for (int n = 0; n < 8; ++n) acc[i][n] = {};

#pragma unroll 1
  for (int ks = 0; ks < 16; ++ks) {
    const int k0 = ks * 32;
    // A fragments (16x32 bf16): lanes 0-15 hold K {0..7,16..23}, lanes 16-31
    // hold K {8..15,24..31} of row M = lane&15.
    v16bf a[2];
#pragma unroll
    for (int mbi = 0; mbi < 2; ++mbi) {
      const int m = (wave * 2 + mbi) * 16 + nidx;
      const __bf16* ap = qnb + (size_t)m * DIM + k0 + half * 8;
      const v8bf lo = *(const v8bf*)ap;
      const v8bf hi = *(const v8bf*)(ap + 16);
#pragma unroll
      for (int e = 0; e < 8; ++e) { a[mbi][e] = lo[e]; a[mbi][8 + e] = hi[e]; }
    }
#pragma unroll
    for (int nb = 0; nb < 8; ++nb) {
      // B fragment (32x16 bf16): lanes 0-15 hold K {0..7,16..23}, lanes 16-31
      // hold K {8..15,24..31} of column n = lane&15.
      const __bf16* bp = &pB[nb * 16 + nidx][k0 + half * 8];
      const v8bf blo = *(const v8bf*)bp;
      const v8bf bhi = *(const v8bf*)(bp + 16);
      v16bf b;
#pragma unroll
      for (int e = 0; e < 8; ++e) { b[e] = blo[e]; b[8 + e] = bhi[e]; }
      acc[0][nb] = __builtin_amdgcn_wmma_f32_16x16x32_bf16(
          false, a[0], false, b, (short)0, acc[0][nb], false, false);
      acc[1][nb] = __builtin_amdgcn_wmma_f32_16x16x32_bf16(
          false, a[1], false, b, (short)0, acc[1][nb], false, false);
    }
  }

  // ---- scale by 1/||p|| and store sims tile ----
#pragma unroll
  for (int mbi = 0; mbi < 2; ++mbi) {
    const int mb = wave * 2 + mbi;
#pragma unroll
    for (int nb = 0; nb < 8; ++nb) {
      const float rn = rnorm[nb * 16 + nidx];
      const size_t ncol = pat0 + nb * 16 + nidx;
#pragma unroll
      for (int r = 0; r < 8; ++r) {
        // C layout: VGPR r, lanes 0-15 -> M=r, lanes 16-31 -> M=8+r; N=lane&15
        const int qrow = mb * 16 + half * 8 + r;
        sims[(size_t)qrow * N_PAT + ncol] = acc[mbi][nb][r] * rn;
      }
    }
  }
}

// ---------------------------------------------------------------------------
// Kernel 3: per-query top-32 over 262144 sims.
// One WG (256 thr) per query. Each wave streams its 32768-sim slice through
// a wave-private double-buffered LDS stage filled by
// GLOBAL_LOAD_ASYNC_TO_LDS_B128 (ASYNCcnt, s_wait_asynccnt pipelining).
// Each thread keeps a local top-32 in LDS via rare min-replacement, then
// 32 rounds of shuffle+LDS argmax merge across the 8192 candidates.
// ---------------------------------------------------------------------------
__global__ __launch_bounds__(256) void topk_kernel(const float* __restrict__ sims,
                                                   float* __restrict__ top_scores,
                                                   int* __restrict__ top_idx) {
  __shared__ float cval[256 * TOPK];                 // 32 KB
  __shared__ int   cidx[256 * TOPK];                 // 32 KB
  __shared__ __align__(16) float stage[8][2][128];   // 8 KB wave-private staging
  __shared__ float wv[8];
  __shared__ int   wk[8];

  const int t = threadIdx.x, q = blockIdx.x;
  const int lane = t & 31, wave = t >> 5;
  const float NEG = -3.402823466e38f;

#pragma unroll
  for (int s = 0; s < TOPK; ++s) cval[t * TOPK + s] = NEG;
  float curMin = NEG;
  int minSlot = 0;

  const float* base = sims + (size_t)q * N_PAT + (size_t)wave * 32768;
  const unsigned long long gbase = (unsigned long long)(uintptr_t)base;
  // workgroup-relative LDS byte offsets of this lane's staging slots
  const unsigned lds0 = (unsigned)(uintptr_t)&stage[wave][0][lane * 4];
  const unsigned lds1 = (unsigned)(uintptr_t)&stage[wave][1][lane * 4];

  // prologue: async-stage chunk 0 into buf 0
  async_g2l_b128(lds0, gbase + (unsigned long long)lane * 16ull);

  for (int i = 0; i < 256; ++i) {
    const int buf = i & 1;
    if (i + 1 < 256) {  // keep one chunk in flight
      async_g2l_b128(buf ? lds0 : lds1,
                     gbase + ((unsigned long long)(i + 1) * 128ull +
                              (unsigned long long)lane * 4ull) * 4ull);
      wait_asynccnt_le1();  // chunk i landed; chunk i+1 still in flight
    } else {
      wait_asynccnt_0();
    }
#pragma unroll
    for (int j = 0; j < 4; ++j) {
      const float v = stage[wave][buf][lane * 4 + j];
      if (v > curMin) {  // rare path: replace current min, rescan for new min
        const int gi = wave * 32768 + i * 128 + lane * 4 + j;
        cval[t * TOPK + minSlot] = v;
        cidx[t * TOPK + minSlot] = gi;
        float nm = 3.402823466e38f;
        int ns = 0;
#pragma unroll
        for (int s = 0; s < TOPK; ++s) {
          const float x = cval[t * TOPK + s];
          if (x < nm) { nm = x; ns = s; }
        }
        curMin = nm;
        minSlot = ns;
      }
    }
  }
  __syncthreads();

  // merge: 32 rounds of block-wide argmax over the 8192 candidates
  for (int r = 0; r < TOPK; ++r) {
    float bv = NEG;
    int bk = t * TOPK;
#pragma unroll
    for (int s = 0; s < TOPK; ++s) {
      const float x = cval[t * TOPK + s];
      if (x > bv) { bv = x; bk = t * TOPK + s; }
    }
#pragma unroll
    for (int m = 16; m > 0; m >>= 1) {
      const float ov = __shfl_xor(bv, m, 32);
      const int ok = __shfl_xor(bk, m, 32);
      if (ov > bv) { bv = ov; bk = ok; }
    }
    if (lane == 0) { wv[wave] = bv; wk[wave] = bk; }
    __syncthreads();
    if (t == 0) {
      float fbv = wv[0];
      int fbk = wk[0];
#pragma unroll
      for (int i = 1; i < 8; ++i)
        if (wv[i] > fbv) { fbv = wv[i]; fbk = wk[i]; }
      top_scores[q * TOPK + r] = fbv;
      top_idx[q * TOPK + r] = cidx[fbk];
      cval[fbk] = NEG;  // consume winner
    }
    __syncthreads();
  }
}

// ---------------------------------------------------------------------------
// Kernel 4: retrieved[q] = sum_k softmax(top_scores[q])_k * mem[top_idx[q,k]]
// ---------------------------------------------------------------------------
__global__ __launch_bounds__(256) void aggregate_kernel(
    const float* __restrict__ mem, const float* __restrict__ top_scores,
    const int* __restrict__ top_idx, float* __restrict__ out) {
  __shared__ float wts[TOPK];
  __shared__ int idx[TOPK];
  const int t = threadIdx.x, q = blockIdx.x;
  if (t == 0) {
    float s[TOPK];
    float m = -3.402823466e38f;
#pragma unroll
    for (int k = 0; k < TOPK; ++k) {
      s[k] = top_scores[q * TOPK + k];
      m = fmaxf(m, s[k]);
    }
    float sum = 0.f;
#pragma unroll
    for (int k = 0; k < TOPK; ++k) { s[k] = expf(s[k] - m); sum += s[k]; }
    const float inv = 1.0f / sum;
#pragma unroll
    for (int k = 0; k < TOPK; ++k) wts[k] = s[k] * inv;
  }
  if (t < TOPK) idx[t] = top_idx[q * TOPK + t];
  __syncthreads();
  float a0 = 0.f, a1 = 0.f;
#pragma unroll 4
  for (int k = 0; k < TOPK; ++k) {
    const float* p = mem + (size_t)idx[k] * DIM;
    a0 = fmaf(wts[k], p[t], a0);
    a1 = fmaf(wts[k], p[t + 256], a1);
  }
  out[(size_t)q * DIM + t] = a0;
  out[(size_t)q * DIM + 256 + t] = a1;
}

// ---------------------------------------------------------------------------
extern "C" void kernel_launch(void* const* d_in, const int* in_sizes, int n_in,
                              void* d_out, int out_size, void* d_ws, size_t ws_size,
                              hipStream_t stream) {
  const float* query = (const float*)d_in[0];  // [256, 512] fp32
  const float* mem   = (const float*)d_in[1];  // [262144, 512] fp32
  // d_in[2] = top_k scalar (fixed 32 here)

  float* out = (float*)d_out;                     // retrieved[256*512] ++ scores[256*32]
  char* ws = (char*)d_ws;
  __bf16* qnb  = (__bf16*)ws;                                       // 256 KB
  float*  sims = (float*)(ws + (size_t)(1 << 20));                  // 256 MB
  int*    tidx = (int*)(ws + (size_t)(1 << 20) +
                        (size_t)BQ * N_PAT * sizeof(float));        // 32 KB
  float* tscores = out + (size_t)BQ * DIM;

  qnorm_kernel<<<BQ, 256, 0, stream>>>(query, qnb);
  sims_wmma_kernel<<<N_PAT / NT, 256, 0, stream>>>(mem, qnb, sims);
  topk_kernel<<<BQ, 256, 0, stream>>>(sims, tscores, tidx);
  aggregate_kernel<<<BQ, 256, 0, stream>>>(mem, tscores, tidx, out);
}
